// MultiheadAttention_28252294873731
// MI455X (gfx1250) — compile-verified, matrix operand order checked by reading, not run on hardware
//
#include <hip/hip_runtime.h>
#include <hip/hip_bf16.h>

// ---- problem constants ------------------------------------------------------
#define BB 8
#define SS 1024
#define EE 1152
#define HH 16
#define DD 72
#define DP 96          // head dim padded to 96 = 3 * 32 (exact WMMA K steps)
#define N3 3456        // 3*E

typedef __bf16 bf16_t;
typedef __attribute__((ext_vector_type(16))) __bf16 v16bf;
typedef __attribute__((ext_vector_type(8)))  __bf16 v8bf;
typedef __attribute__((ext_vector_type(8)))  float  v8f;
typedef __attribute__((ext_vector_type(4)))  int    v4i;

// ---- CDNA5 async global->LDS copy (ASYNCcnt path) ---------------------------
#if __has_builtin(__builtin_amdgcn_global_load_async_to_lds_b128)
#define USE_ASYNC_LDS 1
#else
#define USE_ASYNC_LDS 0
#endif

#if USE_ASYNC_LDS
typedef __attribute__((address_space(1))) v4i gas_v4i;   // global int4
typedef __attribute__((address_space(3))) v4i las_v4i;   // LDS int4
#endif

__device__ __forceinline__ void copy16_g2lds(void* lds_dst, const void* gsrc) {
#if USE_ASYNC_LDS
    // Build AS pointers from integers: global AS1 addresses are flat addresses;
    // flat pointers to LDS carry the wave-relative LDS byte offset in the low
    // 32 bits (aperture scheme), which is what the instruction's VDST expects.
    __builtin_amdgcn_global_load_async_to_lds_b128(
        (gas_v4i*)(unsigned long long)(size_t)gsrc,
        (las_v4i*)(unsigned int)(size_t)lds_dst,
        0, 0);
#else
    *(v8bf*)lds_dst = *(const v8bf*)gsrc;
#endif
}

__device__ __forceinline__ void wait_async_copies() {
#if USE_ASYNC_LDS
    asm volatile("s_wait_asynccnt 0x0" ::: "memory");
#endif
}

// ---- WMMA helpers -----------------------------------------------------------
// 16-bit operand lane layout (ISA 7.12.2): lane holds row (M or N) = lane&15,
// K elements e<8 at K = 8*half + e, e>=8 at K = 16 + 8*half + (e-8).
// => two contiguous 8-element (16B) runs at byte offsets 16*half and 32+16*half.
__device__ __forceinline__ v16bf ldsFrag(const bf16_t* base, int row0, int stride, int lane) {
    const int r    = row0 + (lane & 15);
    const int half = lane >> 4;
    const bf16_t* p = base + r * stride + half * 8;
    v8bf lo = *(const v8bf*)(p);
    v8bf hi = *(const v8bf*)(p + 16);
    v16bf f;
#pragma unroll
    for (int i = 0; i < 8; ++i) { f[i] = lo[i]; f[i + 8] = hi[i]; }
    return f;
}

__device__ __forceinline__ v8f wmma32(v16bf a, v16bf b, v8f c) {
    // D = A(16x32 bf16) * B(32x16 bf16) + C(16x16 f32)
    return __builtin_amdgcn_wmma_f32_16x16x32_bf16(
        /*neg_a=*/false, a, /*neg_b=*/false, b,
        /*c_mod=*/(short)0, c, /*reuse_a=*/false, /*reuse_b=*/false);
}

// ---- kernel 0: zero the D padding of Qp/Kp/Vp -------------------------------
__global__ __launch_bounds__(256) void zero_pad_kernel(bf16_t* __restrict__ Qp,
                                                       bf16_t* __restrict__ Kp,
                                                       bf16_t* __restrict__ Vp) {
    const int idx = blockIdx.x * 256 + threadIdx.x;   // B*H*S = 131072 rows
    v8bf z = {};
    bf16_t* q = Qp + (size_t)idx * DP + DD;
    bf16_t* k = Kp + (size_t)idx * DP + DD;
    bf16_t* v = Vp + (size_t)idx * DP + DD;
#pragma unroll
    for (int i = 0; i < 3; ++i) {                     // 24 bf16 = 3 x 16B
        *(v8bf*)(q + i * 8) = z;
        *(v8bf*)(k + i * 8) = z;
        *(v8bf*)(v + i * 8) = z;
    }
}

// ---- kernel 1: fused QKV projection, scatter into [B,H,S,DP] bf16 -----------
__global__ __launch_bounds__(256) void qkv_proj_kernel(
    const float* __restrict__ q, const float* __restrict__ kin, const float* __restrict__ vin,
    const float* __restrict__ W, const float* __restrict__ bias,
    bf16_t* __restrict__ Qp, bf16_t* __restrict__ Kp, bf16_t* __restrict__ Vp) {
    __shared__ bf16_t As[128 * 40];   // 128 x 32 (stride 40 to spread banks)
    __shared__ bf16_t Bs[64 * 40];    // B^T: 64 x 32

    const int n0 = blockIdx.x * 64;   // within [0, 3456); 64 | 1152 so one third only
    const int m0 = blockIdx.y * 128;  // within [0, 8192)
    const int which = n0 / EE;        // 0=q, 1=k, 2=v
    const float* X = (which == 0) ? q : ((which == 1) ? kin : vin);

    const int tid = threadIdx.x, lane = tid & 31, w = tid >> 5;
    const int wm = w >> 1, wn = w & 1;   // 4x2 wave grid -> 32x32 per wave
    v8f acc[2][2] = {};

    for (int k0 = 0; k0 < EE; k0 += 32) {
        // stage A: 128x32 f32 -> bf16
#pragma unroll
        for (int i = 0; i < 4; ++i) {
            int lin = tid + i * 256;
            int row = lin >> 3, c4 = lin & 7;
            const float4 f = *(const float4*)(X + (size_t)(m0 + row) * EE + k0 + c4 * 4);
            bf16_t* d = As + row * 40 + c4 * 4;
            d[0] = (bf16_t)f.x; d[1] = (bf16_t)f.y; d[2] = (bf16_t)f.z; d[3] = (bf16_t)f.w;
        }
        // stage B^T: W[k0..k0+32, n0..n0+64] -> Bs[n][k]
#pragma unroll
        for (int i = 0; i < 2; ++i) {
            int lin = tid + i * 256;
            int kk = lin >> 4, n4 = lin & 15;
            const float4 f = *(const float4*)(W + (size_t)(k0 + kk) * N3 + n0 + n4 * 4);
            Bs[(n4 * 4 + 0) * 40 + kk] = (bf16_t)f.x;
            Bs[(n4 * 4 + 1) * 40 + kk] = (bf16_t)f.y;
            Bs[(n4 * 4 + 2) * 40 + kk] = (bf16_t)f.z;
            Bs[(n4 * 4 + 3) * 40 + kk] = (bf16_t)f.w;
        }
        if (k0 + 32 < EE) {
            __builtin_prefetch(X + (size_t)(m0 + (tid >> 1)) * EE + k0 + 32, 0, 0);
            __builtin_prefetch(W + (size_t)(k0 + 32 + (tid & 31)) * N3 + n0, 0, 0);
        }
        __syncthreads();

        v16bf a0 = ldsFrag(As, wm * 32,      40, lane);
        v16bf a1 = ldsFrag(As, wm * 32 + 16, 40, lane);
        v16bf b0 = ldsFrag(Bs, wn * 32,      40, lane);
        v16bf b1 = ldsFrag(Bs, wn * 32 + 16, 40, lane);
        acc[0][0] = wmma32(a0, b0, acc[0][0]);
        acc[0][1] = wmma32(a0, b1, acc[0][1]);
        acc[1][0] = wmma32(a1, b0, acc[1][0]);
        acc[1][1] = wmma32(a1, b1, acc[1][1]);
        __syncthreads();
    }

    // epilogue: +bias, scatter bf16 into head-major padded layout
    bf16_t* dst = (which == 0) ? Qp : ((which == 1) ? Kp : Vp);
    const int half = lane >> 4, nl = lane & 15;
#pragma unroll
    for (int i = 0; i < 2; ++i)
#pragma unroll
        for (int j = 0; j < 2; ++j) {
            const int ng = n0 + wn * 32 + j * 16 + nl;
            const float bval = bias[ng];
            const int e = ng % EE;
            const int h = e / DD, d = e % DD;
#pragma unroll
            for (int r = 0; r < 8; ++r) {
                const int m = m0 + wm * 32 + i * 16 + r + half * 8;
                const int b = m >> 10, s = m & 1023;
                const float val = acc[i][j][r] + bval;
                dst[(size_t)((b * HH + h) * SS + s) * DP + d] = (bf16_t)val;
            }
        }
}

// ---- kernel 2: flash attention, 64 queries per WG, 64-key chunks ------------
__global__ __launch_bounds__(256) void attn_kernel(
    const bf16_t* __restrict__ Qp, const bf16_t* __restrict__ Kp,
    const bf16_t* __restrict__ Vp, bf16_t* __restrict__ Att) {
    __shared__ bf16_t Qs[64 * 96];
    __shared__ bf16_t Ks[64 * 96];
    __shared__ bf16_t Vt[96 * 64];        // V^T : [d][key]
    __shared__ float  Sc[64 * 65];        // raw scores (f32), stride 65 (bank-friendly)
    __shared__ bf16_t Ps[64 * 72];        // exp probs (bf16), stride 72 (16B aligned)
    __shared__ float  mrow[64], lrow[64], arow[64];

    const int q0 = blockIdx.x * 64;
    const int bh = blockIdx.y;
    const int b = bh / HH, h = bh % HH;
    const int tid = threadIdx.x, lane = tid & 31, w = tid >> 5;

    const bf16_t* Qg = Qp + (size_t)(bh * SS + q0) * DP;
#pragma unroll
    for (int i = 0; i < 3; ++i) {         // 64*96 bf16 = 768 x 16B, async -> LDS
        int lin = tid + i * 256;
        copy16_g2lds(Qs + lin * 8, Qg + lin * 8);
    }
    if (tid < 64) { mrow[tid] = -__builtin_inff(); lrow[tid] = 0.f; }

    v8f o[3] = {};                        // running O: wave owns 16 rows x 48 d-cols
    const int om = w & 3;                 // O M-tile (rows om*16..)
    const int oh = w >> 2;                // d half: cols oh*48..
    const float scale = 0.11785113019775792f;  // 1/sqrt(72)

    for (int kb = 0; kb < SS; kb += 64) {
        const bf16_t* Kg = Kp + (size_t)(bh * SS + kb) * DP;
        const bf16_t* Vg = Vp + (size_t)(bh * SS + kb) * DP;
#pragma unroll
        for (int i = 0; i < 3; ++i) {     // K chunk: async -> LDS
            int lin = tid + i * 256;
            copy16_g2lds(Ks + lin * 8, Kg + lin * 8);
        }
#pragma unroll
        for (int i = 0; i < 3; ++i) {     // V chunk: b128 read, bf16 scatter-transpose
            int c = tid + i * 256;        // 768 chunks of 8 bf16
            int kk = c / 12, d0 = (c % 12) * 8;
            v8bf vv = *(const v8bf*)(Vg + kk * DP + d0);
#pragma unroll
            for (int j = 0; j < 8; ++j) Vt[(d0 + j) * 64 + kk] = vv[j];
        }
        wait_async_copies();
        __syncthreads();

        // ---- scores S = (Q * scale) . K^T : 64x64, wave = 16 rows x 32 cols
        {
            const int sm = w >> 1;
            const int sn = (w & 1) * 32;
            v8f sacc[2] = {};
#pragma unroll
            for (int ks = 0; ks < DP; ks += 32) {
                v16bf aq = ldsFrag(Qs + ks, sm * 16, 96, lane);
                v16bf b0 = ldsFrag(Ks + ks, sn,      96, lane);
                v16bf b1 = ldsFrag(Ks + ks, sn + 16, 96, lane);
                sacc[0] = wmma32(aq, b0, sacc[0]);
                sacc[1] = wmma32(aq, b1, sacc[1]);
            }
            const int half = lane >> 4, nl = lane & 15;
#pragma unroll
            for (int j = 0; j < 2; ++j)
#pragma unroll
                for (int r = 0; r < 8; ++r) {
                    const int row = sm * 16 + r + half * 8;
                    const int col = sn + j * 16 + nl;
                    Sc[row * 65 + col] = sacc[j][r] * scale;
                }
        }
        __syncthreads();

        // ---- online softmax per row (one thread per row)
        if (tid < 64) {
            const int row = tid;
            const float* sr = Sc + row * 65;
            float cm = mrow[row];
            for (int c = 0; c < 64; ++c) cm = fmaxf(cm, sr[c]);
            const float alpha = __expf(mrow[row] - cm);
            float sum = 0.f;
            bf16_t* pr = Ps + row * 72;
            for (int c = 0; c < 64; ++c) {
                const float p = __expf(sr[c] - cm);
                sum += p;
                pr[c] = (bf16_t)p;
            }
            lrow[row] = lrow[row] * alpha + sum;
            mrow[row] = cm;
            arow[row] = alpha;
        }
        __syncthreads();

        // ---- O = alpha*O + P . V
        {
            const int half = lane >> 4;
            float al[8];
#pragma unroll
            for (int r = 0; r < 8; ++r) al[r] = arow[om * 16 + r + half * 8];
#pragma unroll
            for (int t = 0; t < 3; ++t)
#pragma unroll
                for (int r = 0; r < 8; ++r) o[t][r] *= al[r];
#pragma unroll
            for (int ks = 0; ks < 64; ks += 32) {
                v16bf ap = ldsFrag(Ps + ks, om * 16, 72, lane);
#pragma unroll
                for (int t = 0; t < 3; ++t) {
                    v16bf bv = ldsFrag(Vt + ks, oh * 48 + t * 16, 64, lane);
                    o[t] = wmma32(ap, bv, o[t]);
                }
            }
        }
        __syncthreads();
    }

    // ---- normalize and write concat-head bf16 output [B,S,E]
    const int half = lane >> 4, nl = lane & 15;
#pragma unroll
    for (int t = 0; t < 3; ++t)
#pragma unroll
        for (int r = 0; r < 8; ++r) {
            const int row = om * 16 + r + half * 8;
            const int d = oh * 48 + t * 16 + nl;
            if (d < DD) {
                const float val = o[t][r] / lrow[row];
                Att[(size_t)(b * SS + q0 + row) * EE + h * DD + d] = (bf16_t)val;
            }
        }
}

// ---- kernel 3: output projection, f32 result + bias -------------------------
__global__ __launch_bounds__(256) void out_proj_kernel(
    const bf16_t* __restrict__ Att, const float* __restrict__ Wo,
    const float* __restrict__ bo, float* __restrict__ out) {
    __shared__ bf16_t As[128 * 40];
    __shared__ bf16_t Bs[64 * 40];

    const int n0 = blockIdx.x * 64;
    const int m0 = blockIdx.y * 128;
    const int tid = threadIdx.x, lane = tid & 31, w = tid >> 5;
    const int wm = w >> 1, wn = w & 1;
    v8f acc[2][2] = {};

    for (int k0 = 0; k0 < EE; k0 += 32) {
#pragma unroll
        for (int i = 0; i < 2; ++i) {                   // A already bf16: async -> LDS
            int lin = tid + i * 256;
            int row = lin >> 2, c8 = lin & 3;
            copy16_g2lds(As + row * 40 + c8 * 8,
                         Att + (size_t)(m0 + row) * EE + k0 + c8 * 8);
        }
#pragma unroll
        for (int i = 0; i < 2; ++i) {                   // W^T stage with f32->bf16
            int lin = tid + i * 256;
            int kk = lin >> 4, n4 = lin & 15;
            const float4 f = *(const float4*)(Wo + (size_t)(k0 + kk) * EE + n0 + n4 * 4);
            Bs[(n4 * 4 + 0) * 40 + kk] = (bf16_t)f.x;
            Bs[(n4 * 4 + 1) * 40 + kk] = (bf16_t)f.y;
            Bs[(n4 * 4 + 2) * 40 + kk] = (bf16_t)f.z;
            Bs[(n4 * 4 + 3) * 40 + kk] = (bf16_t)f.w;
        }
        if (k0 + 32 < EE)
            __builtin_prefetch(Att + (size_t)(m0 + (tid >> 1)) * EE + k0 + 32, 0, 0);
        wait_async_copies();
        __syncthreads();

        v16bf a0 = ldsFrag(As, wm * 32,      40, lane);
        v16bf a1 = ldsFrag(As, wm * 32 + 16, 40, lane);
        v16bf b0 = ldsFrag(Bs, wn * 32,      40, lane);
        v16bf b1 = ldsFrag(Bs, wn * 32 + 16, 40, lane);
        acc[0][0] = wmma32(a0, b0, acc[0][0]);
        acc[0][1] = wmma32(a0, b1, acc[0][1]);
        acc[1][0] = wmma32(a1, b0, acc[1][0]);
        acc[1][1] = wmma32(a1, b1, acc[1][1]);
        __syncthreads();
    }

    const int half = lane >> 4, nl = lane & 15;
#pragma unroll
    for (int i = 0; i < 2; ++i)
#pragma unroll
        for (int j = 0; j < 2; ++j) {
            const int ng = n0 + wn * 32 + j * 16 + nl;
            const float bval = bo[ng];
#pragma unroll
            for (int r = 0; r < 8; ++r) {
                const int m = m0 + wm * 32 + i * 16 + r + half * 8;
                out[(size_t)m * EE + ng] = acc[i][j][r] + bval;
            }
        }
}

// ---- host launcher ----------------------------------------------------------
extern "C" void kernel_launch(void* const* d_in, const int* in_sizes, int n_in,
                              void* d_out, int out_size, void* d_ws, size_t ws_size,
                              hipStream_t stream) {
    const float* q    = (const float*)d_in[0];
    const float* k    = (const float*)d_in[1];
    const float* v    = (const float*)d_in[2];
    const float* W    = (const float*)d_in[3];
    const float* bias = (const float*)d_in[4];
    const float* Wo   = (const float*)d_in[5];
    const float* bo   = (const float*)d_in[6];
    float* out        = (float*)d_out;

    const size_t qkvElems = (size_t)BB * HH * SS * DP;           // 12.58M bf16 each
    bf16_t* Qp  = (bf16_t*)d_ws;
    bf16_t* Kp  = Qp + qkvElems;
    bf16_t* Vp  = Kp + qkvElems;
    bf16_t* Att = Vp + qkvElems;                                 // [B,S,E] bf16

    zero_pad_kernel<<<dim3((BB * HH * SS) / 256), dim3(256), 0, stream>>>(Qp, Kp, Vp);
    qkv_proj_kernel<<<dim3(N3 / 64, (BB * SS) / 128), dim3(256), 0, stream>>>(
        q, k, v, W, bias, Qp, Kp, Vp);
    attn_kernel<<<dim3(SS / 64, BB * HH), dim3(256), 0, stream>>>(Qp, Kp, Vp, Att);
    out_proj_kernel<<<dim3(EE / 64, (BB * SS) / 128), dim3(256), 0, stream>>>(
        Att, Wo, bo, out);
}
